// MambaBlock_71657234366969
// MI455X (gfx1250) — matrix operand order repaired and checked by reading, not compiled, not run on hardware
//
#include <hip/hip_runtime.h>
#include <hip/hip_bf16.h>

// ---------------------------------------------------------------------------
// Problem constants (B=2, L=1024, H=2048, DT_RANK=64, N_STATE=16)
// ---------------------------------------------------------------------------
#define BATCH   2
#define LSEQ    1024
#define HDIM    2048
#define MROWS   (BATCH * LSEQ)     // 2048
#define DTRANK  64
#define NSTATE  16
#define DBCW    (DTRANK + 2 * NSTATE)  // 96

// WMMA tile config: block tile 128x128, K-step 32, 256 threads = 8 waves (4x2)
#define TM   128
#define TN   128
#define TK   32
#define LDT  40        // padded LDS row length (elements); 80B rows, 16B aligned

typedef __attribute__((ext_vector_type(16))) __bf16 v16bf;
typedef __attribute__((ext_vector_type(8)))  __bf16 v8bf;
typedef __attribute__((ext_vector_type(8)))  float  v8f;

// ---------------------------------------------------------------------------
// Optional gfx1250 async global->LDS copy path (ASYNCcnt tracked)
// Builtin signature (from clang diagnostic):
//   void __builtin_amdgcn_global_load_async_to_lds_b128(
//       int __vector(4) addrspace(1)* src, int __vector(4) addrspace(3)* dst,
//       imm int offset, imm int cpol)
// ---------------------------------------------------------------------------
#if defined(__has_builtin)
#if __has_builtin(__builtin_amdgcn_global_load_async_to_lds_b128)
#define HAVE_ASYNC_LDS 1
#endif
#endif
#ifndef HAVE_ASYNC_LDS
#define HAVE_ASYNC_LDS 0
#endif

typedef int v4i_gcc __attribute__((vector_size(16)));
typedef __attribute__((address_space(1))) v4i_gcc v4i_as1;
typedef __attribute__((address_space(3))) v4i_gcc v4i_as3;

// Copy 16 bytes (8 bf16) global -> LDS, per lane.
__device__ __forceinline__ void copy16_to_lds(const void* g, void* l) {
#if HAVE_ASYNC_LDS
    __builtin_amdgcn_global_load_async_to_lds_b128((v4i_as1*)g, (v4i_as3*)l, 0, 0);
#else
    *(uint4*)l = *(const uint4*)g;
#endif
}

__device__ __forceinline__ void async_wait() {
#if HAVE_ASYNC_LDS
#if __has_builtin(__builtin_amdgcn_s_wait_asynccnt)
    __builtin_amdgcn_s_wait_asynccnt(0);
#else
    asm volatile("s_wait_asynccnt 0x0" ::: "memory");
#endif
#endif
}

// Assemble a 16-element bf16 fragment from two 16-byte LDS chunks.
__device__ __forceinline__ v16bf frag16(const __bf16* p0, const __bf16* p1) {
    v8bf lo = *(const v8bf*)p0;
    v8bf hi = *(const v8bf*)p1;
    v16bf r;
#pragma unroll
    for (int i = 0; i < 8; ++i) { r[i] = lo[i]; r[i + 8] = hi[i]; }
    return r;
}

__device__ __forceinline__ v8f zero8() {
    v8f z;
#pragma unroll
    for (int i = 0; i < 8; ++i) z[i] = 0.0f;
    return z;
}

// ---------------------------------------------------------------------------
// Weight conversion kernels
// ---------------------------------------------------------------------------
__global__ void cvt_bf16_kernel(const float* __restrict__ src,
                                __bf16* __restrict__ dst, size_t n) {
    size_t i = (size_t)blockIdx.x * blockDim.x + threadIdx.x;
    size_t stride = (size_t)gridDim.x * blockDim.x;
    for (; i < n; i += stride) dst[i] = (__bf16)src[i];
}

// conv_w[k][i][o] (LIO) -> cwT[k][o][i]  (per-tap N x K, bf16)
__global__ void conv_transpose_bf16_kernel(const float* __restrict__ cw,
                                           __bf16* __restrict__ cwT) {
    size_t n = (size_t)4 * HDIM * HDIM;
    size_t idx = (size_t)blockIdx.x * blockDim.x + threadIdx.x;
    size_t stride = (size_t)gridDim.x * blockDim.x;
    for (; idx < n; idx += stride) {
        size_t k = idx / ((size_t)HDIM * HDIM);
        size_t rem = idx % ((size_t)HDIM * HDIM);
        size_t o = rem / HDIM;
        size_t i = rem % HDIM;
        cwT[idx] = (__bf16)cw[k * HDIM * HDIM + i * HDIM + o];
    }
}

// ---------------------------------------------------------------------------
// RMSNorm: one 256-thread block per row of H=2048
// ---------------------------------------------------------------------------
__global__ __launch_bounds__(256)
void rmsnorm_kernel(const float* __restrict__ x, const float* __restrict__ w,
                    float* __restrict__ h) {
    __shared__ float red[256];
    const size_t r = blockIdx.x;
    const float* xr = x + r * HDIM;
    float s = 0.0f;
    for (int i = threadIdx.x; i < HDIM; i += 256) {
        float v = xr[i];
        s += v * v;
    }
    red[threadIdx.x] = s;
    __syncthreads();
    for (int off = 128; off > 0; off >>= 1) {
        if (threadIdx.x < off) red[threadIdx.x] += red[threadIdx.x + off];
        __syncthreads();
    }
    float rstd = rsqrtf(red[0] / (float)HDIM + 1e-5f);
    for (int i = threadIdx.x; i < HDIM; i += 256)
        h[r * HDIM + i] = xr[i] * rstd * w[i];
}

// ---------------------------------------------------------------------------
// Generic WMMA GEMM: C[M,N] = epilogue( A[M,K](f32) @ W[N,K](bf16)^T )
// EPI: 0 = none, 1 = bias + softplus, 2 = residual add
// ---------------------------------------------------------------------------
template <int EPI>
__global__ __launch_bounds__(256)
void gemm_bf16w_kernel(const float* __restrict__ A, int lda,
                       const __bf16* __restrict__ W, int ldw,
                       float* __restrict__ C, int ldc,
                       int M, int N, int K,
                       const float* __restrict__ bias,
                       const float* __restrict__ resid, int ldr) {
    __shared__ __bf16 As[TM][LDT];
    __shared__ __bf16 Ws[TN][LDT];

    const int tid   = threadIdx.x;
    const int lane  = tid & 31;
    const int wave  = tid >> 5;     // 0..7
    const int waveM = wave & 3;     // 4 waves over M
    const int waveN = wave >> 2;    // 2 waves over N
    const int lhalf = lane >> 4;    // 0/1
    const int lm    = lane & 15;
    const int abase = lhalf * 8;    // A-fragment K sub-base per ISA layout

    const int m0 = blockIdx.x * TM;
    const int n0 = blockIdx.y * TN;
    const bool fullN = (n0 + TN <= N);  // block-uniform

    const int ldRow = tid >> 1;          // 0..127
    const int ldSeg = (tid & 1) * 16;    // 0 or 16 (K columns)

    v8f acc[2][4];
#pragma unroll
    for (int mi = 0; mi < 2; ++mi)
#pragma unroll
        for (int ni = 0; ni < 4; ++ni) acc[mi][ni] = zero8();

    for (int k0 = 0; k0 < K; k0 += TK) {
        // A tile (f32 -> bf16)
        {
            const float* src = A + (size_t)(m0 + ldRow) * lda + k0 + ldSeg;
#pragma unroll
            for (int j = 0; j < 4; ++j) {
                float4 v = *(const float4*)(src + 4 * j);
                As[ldRow][ldSeg + 4 * j + 0] = (__bf16)v.x;
                As[ldRow][ldSeg + 4 * j + 1] = (__bf16)v.y;
                As[ldRow][ldSeg + 4 * j + 2] = (__bf16)v.z;
                As[ldRow][ldSeg + 4 * j + 3] = (__bf16)v.w;
            }
        }
        // W tile (bf16, N x K): async fast path for full tiles
        if (fullN) {
            const __bf16* src = W + (size_t)(n0 + ldRow) * ldw + k0 + ldSeg;
            copy16_to_lds(src,     &Ws[ldRow][ldSeg]);
            copy16_to_lds(src + 8, &Ws[ldRow][ldSeg + 8]);
        } else {
            int gn = n0 + ldRow;
            if (gn < N) {
                const __bf16* src = W + (size_t)gn * ldw + k0 + ldSeg;
                *(uint4*)&Ws[ldRow][ldSeg]     = *(const uint4*)src;
                *(uint4*)&Ws[ldRow][ldSeg + 8] = *(const uint4*)(src + 8);
            } else {
                uint4 z = {0u, 0u, 0u, 0u};
                *(uint4*)&Ws[ldRow][ldSeg]     = z;
                *(uint4*)&Ws[ldRow][ldSeg + 8] = z;
            }
        }
        async_wait();
        __syncthreads();

        const int wm = waveM * 32;
        const int wn = waveN * 64;
        v16bf afrag[2], bfrag[4];
#pragma unroll
        for (int mi = 0; mi < 2; ++mi) {
            const __bf16* row = &As[wm + mi * 16 + lm][0];
            afrag[mi] = frag16(row + abase, row + 16 + abase);
        }
#pragma unroll
        for (int ni = 0; ni < 4; ++ni) {
            const __bf16* row = &Ws[wn + ni * 16 + lm][0];
            bfrag[ni] = frag16(row + lhalf * 8, row + 16 + lhalf * 8);
        }
#pragma unroll
        for (int mi = 0; mi < 2; ++mi)
#pragma unroll
            for (int ni = 0; ni < 4; ++ni)
                acc[mi][ni] = __builtin_amdgcn_wmma_f32_16x16x32_bf16(
                    false, afrag[mi], false, bfrag[ni],
                    (short)0, acc[mi][ni], false, false);
        __syncthreads();
    }

    // Epilogue + store (C/D layout: lane lm = N, VGPR v -> M = 8*lhalf + v)
#pragma unroll
    for (int mi = 0; mi < 2; ++mi) {
#pragma unroll
        for (int ni = 0; ni < 4; ++ni) {
            int n = n0 + waveN * 64 + ni * 16 + lm;
            if (n >= N) continue;
#pragma unroll
            for (int v = 0; v < 8; ++v) {
                int m = m0 + waveM * 32 + mi * 16 + lhalf * 8 + v;
                float val = acc[mi][ni][v];
                if (EPI == 1) {  // bias + softplus
                    val += bias[n];
                    val = (val > 20.0f) ? val : log1pf(__expf(val));
                } else if (EPI == 2) {  // residual add
                    val += resid[(size_t)m * ldr + n];
                }
                C[(size_t)m * ldc + n] = val;
            }
        }
    }
}

// ---------------------------------------------------------------------------
// Causal dense conv (kernel=4, pad 3 left) as 4 shifted WMMA GEMMs + bias + SiLU.
// Single shared A tile of TM+3 rows per K-step, reused by all 4 taps.
// A = xc = xz[:, 0:H] (row stride 4096); weights cwT[tap][o][i] (N x K, bf16).
// ---------------------------------------------------------------------------
__global__ __launch_bounds__(256)
void conv_gemm_kernel(const float* __restrict__ xz,
                      const __bf16* __restrict__ cwT,
                      const float* __restrict__ conv_b,
                      float* __restrict__ xconv) {
    __shared__ __bf16 As[TM + 4][LDT];     // rows map to global rows m0-3 .. m0+127
    __shared__ __bf16 Ws[4][TN][LDT];      // one W tile per tap

    const int tid   = threadIdx.x;
    const int lane  = tid & 31;
    const int wave  = tid >> 5;
    const int waveM = wave & 3;
    const int waveN = wave >> 2;
    const int lhalf = lane >> 4;
    const int lm    = lane & 15;
    const int abase = lhalf * 8;

    const int m0 = blockIdx.x * TM;
    const int n0 = blockIdx.y * TN;
    const int batchBase = m0 & ~(LSEQ - 1);  // first row of this batch segment

    const int ldRow = tid >> 1;          // 0..127
    const int ldSeg = (tid & 1) * 16;    // 0 or 16

    v8f acc[2][4];
#pragma unroll
    for (int mi = 0; mi < 2; ++mi)
#pragma unroll
        for (int ni = 0; ni < 4; ++ni) acc[mi][ni] = zero8();

    for (int k0 = 0; k0 < HDIM; k0 += TK) {
        // ---- A tile: rows r=0..130 <-> global rows (m0-3+r), zero before batch ----
        {
            // main 128 rows
            const int gr = m0 - 3 + ldRow;
            if (gr >= batchBase) {
                const float* src = xz + (size_t)gr * (2 * HDIM) + k0 + ldSeg;
#pragma unroll
                for (int j = 0; j < 4; ++j) {
                    float4 v = *(const float4*)(src + 4 * j);
                    As[ldRow][ldSeg + 4 * j + 0] = (__bf16)v.x;
                    As[ldRow][ldSeg + 4 * j + 1] = (__bf16)v.y;
                    As[ldRow][ldSeg + 4 * j + 2] = (__bf16)v.z;
                    As[ldRow][ldSeg + 4 * j + 3] = (__bf16)v.w;
                }
            } else {
#pragma unroll
                for (int j = 0; j < 16; ++j) As[ldRow][ldSeg + j] = (__bf16)0.0f;
            }
            // tail rows 128..130 (always inside batch: m0+125..m0+127)
            if (tid < 6) {
                const int r2 = 128 + (tid >> 1);
                const int seg2 = (tid & 1) * 16;
                const float* src = xz + (size_t)(m0 - 3 + r2) * (2 * HDIM) + k0 + seg2;
#pragma unroll
                for (int j = 0; j < 4; ++j) {
                    float4 v = *(const float4*)(src + 4 * j);
                    As[r2][seg2 + 4 * j + 0] = (__bf16)v.x;
                    As[r2][seg2 + 4 * j + 1] = (__bf16)v.y;
                    As[r2][seg2 + 4 * j + 2] = (__bf16)v.z;
                    As[r2][seg2 + 4 * j + 3] = (__bf16)v.w;
                }
            }
        }
        // ---- W tiles for all 4 taps (async global->LDS) ----
#pragma unroll
        for (int tap = 0; tap < 4; ++tap) {
            const __bf16* src = cwT + (size_t)tap * HDIM * HDIM
                              + (size_t)(n0 + ldRow) * HDIM + k0 + ldSeg;
            copy16_to_lds(src,     &Ws[tap][ldRow][ldSeg]);
            copy16_to_lds(src + 8, &Ws[tap][ldRow][ldSeg + 8]);
        }
        async_wait();
        __syncthreads();

        const int wm = waveM * 32;
        const int wn = waveN * 64;
#pragma unroll
        for (int tap = 0; tap < 4; ++tap) {
            v16bf afrag[2], bfrag[4];
#pragma unroll
            for (int mi = 0; mi < 2; ++mi) {
                // output row m uses source row m + tap - 3 -> local row offset +tap
                const __bf16* row = &As[wm + mi * 16 + lm + tap][0];
                afrag[mi] = frag16(row + abase, row + 16 + abase);
            }
#pragma unroll
            for (int ni = 0; ni < 4; ++ni) {
                const __bf16* row = &Ws[tap][wn + ni * 16 + lm][0];
                bfrag[ni] = frag16(row + lhalf * 8, row + 16 + lhalf * 8);
            }
#pragma unroll
            for (int mi = 0; mi < 2; ++mi)
#pragma unroll
                for (int ni = 0; ni < 4; ++ni)
                    acc[mi][ni] = __builtin_amdgcn_wmma_f32_16x16x32_bf16(
                        false, afrag[mi], false, bfrag[ni],
                        (short)0, acc[mi][ni], false, false);
        }
        __syncthreads();
    }

    // bias + SiLU epilogue
#pragma unroll
    for (int mi = 0; mi < 2; ++mi) {
#pragma unroll
        for (int ni = 0; ni < 4; ++ni) {
            int n = n0 + waveN * 64 + ni * 16 + lm;
#pragma unroll
            for (int v = 0; v < 8; ++v) {
                int m = m0 + waveM * 32 + mi * 16 + lhalf * 8 + v;
                float t = acc[mi][ni][v] + conv_b[n];
                t = t / (1.0f + __expf(-t));  // silu
                xconv[(size_t)m * HDIM + n] = t;
            }
        }
    }
}

// ---------------------------------------------------------------------------
// Selective scan: one thread per (b,h); 16-state recurrence in registers,
// fused with C-contraction, D skip path and SiLU(z) gating.
// ---------------------------------------------------------------------------
__global__ __launch_bounds__(256)
void scan_kernel(const float* __restrict__ delta,   // M x H (softplus'd)
                 const float* __restrict__ dBC,     // M x 96
                 const float* __restrict__ xconv,   // M x H (post conv+silu)
                 const float* __restrict__ xz,      // M x 2H  (z = cols H..2H)
                 const float* __restrict__ A_log,   // H x 16
                 const float* __restrict__ Dp,      // H
                 float* __restrict__ g)             // M x H
{
    const int idx = blockIdx.x * blockDim.x + threadIdx.x;  // 0..B*H-1
    const int b = idx / HDIM;
    const int h = idx % HDIM;

    float Av[NSTATE], s[NSTATE];
#pragma unroll
    for (int n = 0; n < NSTATE; ++n) {
        Av[n] = -__expf(A_log[(size_t)h * NSTATE + n]);
        s[n] = 0.0f;
    }
    const float Dv = Dp[h];

    for (int l = 0; l < LSEQ; ++l) {
        const size_t r = (size_t)b * LSEQ + l;
        const float dlt = delta[r * HDIM + h];
        const float xv  = xconv[r * HDIM + h];
        const float du  = dlt * xv;
        const float* bc = dBC + r * DBCW;
        float y = 0.0f;
#pragma unroll
        for (int n = 0; n < NSTATE; ++n) {
            float dA = __expf(dlt * Av[n]);
            s[n] = dA * s[n] + du * bc[DTRANK + n];
            y += s[n] * bc[DTRANK + NSTATE + n];
        }
        y += Dv * xv;
        const float zv = xz[r * (2 * HDIM) + HDIM + h];
        g[r * HDIM + h] = y * (zv / (1.0f + __expf(-zv)));
    }
}

// ---------------------------------------------------------------------------
// Host launcher
// ---------------------------------------------------------------------------
extern "C" void kernel_launch(void* const* d_in, const int* in_sizes, int n_in,
                              void* d_out, int out_size, void* d_ws, size_t ws_size,
                              hipStream_t stream) {
    const float* x        = (const float*)d_in[0];   // (B,L,H)
    const float* norm_w   = (const float*)d_in[1];   // (H)
    const float* in_proj  = (const float*)d_in[2];   // (2H,H)
    const float* conv_w   = (const float*)d_in[3];   // (4,H,H)
    const float* conv_b   = (const float*)d_in[4];   // (H)
    const float* x_proj   = (const float*)d_in[5];   // (96,H)
    const float* dt_proj  = (const float*)d_in[6];   // (H,64)
    const float* dt_b     = (const float*)d_in[7];   // (H)
    const float* A_log    = (const float*)d_in[8];   // (H,16)
    const float* Dp       = (const float*)d_in[9];   // (H)
    const float* out_proj = (const float*)d_in[10];  // (H,H)
    float* out = (float*)d_out;

    // ---- workspace carve-up (256B aligned) ----
    char* ws = (char*)d_ws;
    size_t off = 0;
    auto carve = [&](size_t bytes) -> char* {
        char* p = ws + off;
        off = (off + bytes + 255) & ~(size_t)255;
        return p;
    };
    __bf16* wpIn   = (__bf16*)carve((size_t)2 * HDIM * HDIM * 2 * sizeof(__bf16));      // 4096x2048
    __bf16* wpConv = (__bf16*)carve((size_t)4 * HDIM * HDIM * sizeof(__bf16));          // 4x2048x2048
    __bf16* wpX    = (__bf16*)carve((size_t)DBCW * HDIM * sizeof(__bf16));              // 96x2048
    __bf16* wpDt   = (__bf16*)carve((size_t)HDIM * DTRANK * sizeof(__bf16));            // 2048x64
    __bf16* wpOut  = (__bf16*)carve((size_t)HDIM * HDIM * sizeof(__bf16));              // 2048x2048
    float*  hbuf   = (float*)carve((size_t)MROWS * HDIM * sizeof(float));               // 2048x2048
    float*  xzbuf  = (float*)carve((size_t)MROWS * 2 * HDIM * sizeof(float));           // 2048x4096
    float*  xconv  = (float*)carve((size_t)MROWS * HDIM * sizeof(float));
    float*  dBC    = (float*)carve((size_t)MROWS * DBCW * sizeof(float));
    float*  delta  = (float*)carve((size_t)MROWS * HDIM * sizeof(float));
    float*  gbuf   = (float*)carve((size_t)MROWS * HDIM * sizeof(float));

    // ---- weight conversion ----
    {
        size_t n;
        n = (size_t)2 * HDIM * HDIM;
        cvt_bf16_kernel<<<(unsigned)((n + 255) / 256), 256, 0, stream>>>(in_proj, wpIn, n);
        n = (size_t)DBCW * HDIM;
        cvt_bf16_kernel<<<(unsigned)((n + 255) / 256), 256, 0, stream>>>(x_proj, wpX, n);
        n = (size_t)HDIM * DTRANK;
        cvt_bf16_kernel<<<(unsigned)((n + 255) / 256), 256, 0, stream>>>(dt_proj, wpDt, n);
        n = (size_t)HDIM * HDIM;
        cvt_bf16_kernel<<<(unsigned)((n + 255) / 256), 256, 0, stream>>>(out_proj, wpOut, n);
        n = (size_t)4 * HDIM * HDIM;
        conv_transpose_bf16_kernel<<<(unsigned)((n + 255) / 256), 256, 0, stream>>>(conv_w, wpConv);
    }

    // ---- 1) RMSNorm ----
    rmsnorm_kernel<<<MROWS, 256, 0, stream>>>(x, norm_w, hbuf);

    dim3 blk(256);

    // ---- 2) in_proj: xz[2048,4096] = h @ in_proj^T ----
    gemm_bf16w_kernel<0><<<dim3(MROWS / TM, (2 * HDIM) / TN), blk, 0, stream>>>(
        hbuf, HDIM, wpIn, HDIM, xzbuf, 2 * HDIM,
        MROWS, 2 * HDIM, HDIM, nullptr, nullptr, 0);

    // ---- 3) causal conv (4 taps) + bias + silu -> xconv ----
    conv_gemm_kernel<<<dim3(MROWS / TM, HDIM / TN), blk, 0, stream>>>(
        xzbuf, wpConv, conv_b, xconv);

    // ---- 4) x_proj: dBC[2048,96] = xconv @ x_proj^T (N masked) ----
    gemm_bf16w_kernel<0><<<dim3(MROWS / TM, (DBCW + TN - 1) / TN), blk, 0, stream>>>(
        xconv, HDIM, wpX, HDIM, dBC, DBCW,
        MROWS, DBCW, HDIM, nullptr, nullptr, 0);

    // ---- 5) dt_proj + softplus: delta[2048,2048] ----
    gemm_bf16w_kernel<1><<<dim3(MROWS / TM, HDIM / TN), blk, 0, stream>>>(
        dBC, DBCW, wpDt, DTRANK, delta, HDIM,
        MROWS, HDIM, DTRANK, dt_b, nullptr, 0);

    // ---- 6) selective scan + gating -> g ----
    scan_kernel<<<(BATCH * HDIM) / 256, 256, 0, stream>>>(
        delta, dBC, xconv, xzbuf, A_log, Dp, gbuf);

    // ---- 7) out_proj + residual: out = x + g @ out_proj^T ----
    gemm_bf16w_kernel<2><<<dim3(MROWS / TM, HDIM / TN), blk, 0, stream>>>(
        gbuf, HDIM, wpOut, HDIM, out, HDIM,
        MROWS, HDIM, HDIM, nullptr, x, HDIM);
}
